// DAF_37160057045012
// MI455X (gfx1250) — compile-verified, matrix operand order checked by reading, not run on hardware
//
#include <hip/hip_runtime.h>
#include <hip/hip_bf16.h>
#include <math.h>

// ---------------------------------------------------------------------------
// MI455X / gfx1250 dual-attention fusion block.
// All GEMM-shaped work -> v_wmma_f32_16x16x32_f16 (wave32, 16x16 tiles).
// LDS tiles are staged in f16 (converted once), A and B both consumed with the
// row-major fragment loader (B staged transposed) so fragment reads vectorize
// to ds_load_b128.  PAM attention is flash-style two-pass (no 4096x4096
// matrix in HBM).
// ---------------------------------------------------------------------------

typedef __attribute__((ext_vector_type(16))) _Float16 v16h;
typedef __attribute__((ext_vector_type(8)))  float    v8f;

#define BN_SCALE 0.99999500003749975f   // 1/sqrt(1 + 1e-5), running_var = 1
#define HW 4096
#define NCH 64

__device__ __forceinline__ v8f wmma32(v16h a, v16h b, v8f c) {
  return __builtin_amdgcn_wmma_f32_16x16x32_f16(
      /*neg_a=*/false, a, /*neg_b=*/false, b,
      /*c_mod=*/(short)0, c, /*reuse_a=*/false, /*reuse_b=*/false);
}

// Fragment loader (A: 16x32 row-major [M][K]; also B when staged as [N][K]).
// Lane L<16 -> row L, lanes 16-31 -> +8 K offset.  Per lane the 16 elements
// form two contiguous 8 x f16 (16B) runs -> 2x ds_load_b128 when ld % 8 == 0.
__device__ __forceinline__ v16h frag_row(const _Float16* src, int ld, int lane) {
  const int mr = lane & 15, kb = (lane >> 4) << 3;
  v16h f;
#pragma unroll
  for (int e = 0; e < 16; ++e) {
    const int j = e >> 1;
    const int kk = ((j & 4) << 2) + kb + ((j & 3) << 1) + (e & 1);
    f[e] = src[mr * ld + kk];
  }
  return f;
}

__device__ __forceinline__ void st4h(_Float16* d, float4 v) {
  d[0] = (_Float16)v.x; d[1] = (_Float16)v.y;
  d[2] = (_Float16)v.z; d[3] = (_Float16)v.w;
}

__device__ __forceinline__ float gelu_exact(float v) {
  return 0.5f * v * (1.0f + erff(v * 0.70710678118654752f));
}
__device__ __forceinline__ float sigmoidf_(float v) {
  return 1.0f / (1.0f + __expf(-v));
}

// ---------------------------------------------------------------------------
// CAM: energy = xf @ xf^T (64x64, K=4096), softmax(rowmin trick) -> att_c.
// One block per batch; 8 waves x 2 tiles.  A and B tiles are the same [c][k]
// chunk staged once in f16.
// ---------------------------------------------------------------------------
__global__ void k_cam_energy(const float* __restrict__ x, float* __restrict__ attc) {
  __shared__ _Float16 xt[64 * 40];  // [c][k-chunk(32)+pad]
  __shared__ float E[64 * 68];
  const int tid = threadIdx.x, b = blockIdx.x;
  const int wave = tid >> 5, lane = tid & 31;
  v8f acc0 = {}, acc1 = {};
  for (int k0 = 0; k0 < HW; k0 += 32) {
    for (int f = tid; f < 512; f += 256) {          // 64 rows x 8 float4
      const int c = f >> 3, k4 = f & 7;
      const float4* p4 =
          (const float4*)(x + (((size_t)b * NCH + c) << 12) + k0) + k4;
      st4h(&xt[c * 40 + k4 * 4], *p4);
    }
    if (k0 + 32 < HW && tid < 64)  // pull next K-chunk (global_prefetch_b8)
      __builtin_prefetch(&x[(((size_t)b * NCH + tid) << 12) + k0 + 32], 0, 1);
    __syncthreads();
#pragma unroll
    for (int t2 = 0; t2 < 2; ++t2) {
      const int ti = wave * 2 + t2, tm = ti >> 2, tn = ti & 3;
      v16h af = frag_row(&xt[tm * 16 * 40], 40, lane);
      v16h bf = frag_row(&xt[tn * 16 * 40], 40, lane);  // B = xf^T tile
      v8f a = t2 ? acc1 : acc0;
      a = wmma32(af, bf, a);
      if (t2) acc1 = a; else acc0 = a;
    }
    __syncthreads();
  }
  const int nr = lane & 15, mb2 = (lane >> 4) << 3;
#pragma unroll
  for (int t2 = 0; t2 < 2; ++t2) {
    const int ti = wave * 2 + t2, tm = ti >> 2, tn = ti & 3;
    v8f a = t2 ? acc1 : acc0;
#pragma unroll
    for (int r = 0; r < 8; ++r)
      E[(tm * 16 + mb2 + r) * 68 + tn * 16 + nr] = a[r];
  }
  __syncthreads();
  if (tid < 64) {
    // softmax(max_d(e) - e) == exp(min_d(e) - e) / sum
    float emin = 3.0e38f;
    for (int d = 0; d < 64; ++d) emin = fminf(emin, E[tid * 68 + d]);
    float s = 0.f;
    for (int d = 0; d < 64; ++d) s += __expf(emin - E[tid * 68 + d]);
    const float inv = 1.0f / s;
    for (int d = 0; d < 64; ++d)
      attc[((size_t)b << 12) + tid * 64 + d] = __expf(emin - E[tid * 68 + d]) * inv;
  }
}

// ---------------------------------------------------------------------------
// Generic 64-out-channel GEMM: out[b,co,n] = act(bn(oscale*A@X + bias)) + resid
// A: [64 x K] (optionally per-batch), X channels from xa (+xb for the 128-ch
// concat).  A staged [co][k] f16, X staged transposed [n][k] f16.
// ---------------------------------------------------------------------------
__global__ void k_gemm64(const float* __restrict__ A, int aBatchStride, int K,
                         const float* __restrict__ xa, const float* __restrict__ xb,
                         const float* __restrict__ bias,
                         const float* __restrict__ bn_g, const float* __restrict__ bn_b,
                         const float* __restrict__ resid,
                         const float* __restrict__ oscale_ptr,
                         float* __restrict__ out, int act) {
  __shared__ _Float16 ldsA[64 * 136];   // [co][K+8]
  __shared__ _Float16 ldsXT[64 * 136];  // [n][K+8]
  const int tid = threadIdx.x;
  const int b = blockIdx.x >> 6;
  const int n0 = (blockIdx.x & 63) << 6;
  const int ld = K + 8;                 // 72 or 136, multiple of 8
  const float oscale = oscale_ptr ? oscale_ptr[0] : 1.0f;
  const float* Ab = A + (size_t)b * aBatchStride;
  // stage A [64][K] via float4
  {
    const int kqs = (K == 128) ? 5 : 4;          // log2(K/4)
    const int kqm = (K >> 2) - 1;
    for (int f = tid; f < (K << 4); f += 256) {  // 64*K/4 float4
      const int row = f >> kqs, c4 = f & kqm;
      const float4* p4 = (const float4*)(Ab + (size_t)row * K) + c4;
      st4h(&ldsA[row * ld + c4 * 4], *p4);
    }
  }
  // stage X transposed [n][k]
  for (int f = tid; f < (K << 4); f += 256) {    // K*64/4 float4 over n
    const int kk = f >> 4, n4 = f & 15;
    const float* src = (kk < 64) ? xa : xb;
    const float4* p4 =
        (const float4*)(src + (((size_t)b * NCH + (kk & 63)) << 12) + n0) + n4;
    const float4 vv = *p4;
    ldsXT[(n4 * 4 + 0) * ld + kk] = (_Float16)vv.x;
    ldsXT[(n4 * 4 + 1) * ld + kk] = (_Float16)vv.y;
    ldsXT[(n4 * 4 + 2) * ld + kk] = (_Float16)vv.z;
    ldsXT[(n4 * 4 + 3) * ld + kk] = (_Float16)vv.w;
  }
  __syncthreads();
  const int wave = tid >> 5, lane = tid & 31;
  const int nr = lane & 15, mb2 = (lane >> 4) << 3;
#pragma unroll
  for (int t2 = 0; t2 < 2; ++t2) {
    const int ti = wave * 2 + t2, tm = ti >> 2, tn = ti & 3;
    v8f acc = {};
    for (int ks = 0; ks < K; ks += 32) {
      v16h af = frag_row(&ldsA[tm * 16 * ld + ks], ld, lane);
      v16h bf = frag_row(&ldsXT[tn * 16 * ld + ks], ld, lane);
      acc = wmma32(af, bf, acc);
    }
#pragma unroll
    for (int r = 0; r < 8; ++r) {
      const int co = tm * 16 + mb2 + r;
      const int n = n0 + tn * 16 + nr;
      float v = acc[r] * oscale + (bias ? bias[co] : 0.f);
      if (bn_g) v = v * (bn_g[co] * BN_SCALE) + bn_b[co];
      if (act == 1) v = fmaxf(v, 0.f);
      else if (act == 2) v = gelu_exact(v);
      const size_t oi = (((size_t)b * NCH + co) << 12) + n;
      if (resid) v += resid[oi];
      out[oi] = v;
    }
  }
}

// ---------------------------------------------------------------------------
// PAM q/k 1x1 convs (C -> 8): tiny K, VALU dot products (thread per pixel).
// ---------------------------------------------------------------------------
__global__ void k_pam_qk(const float* __restrict__ x,
                         const float* __restrict__ wq, const float* __restrict__ bq,
                         const float* __restrict__ wk, const float* __restrict__ bk,
                         float* __restrict__ q, float* __restrict__ k) {
  const int g = blockIdx.x * 256 + threadIdx.x;
  const int b = g >> 12, n = g & 4095;
  float xv[64];
#pragma unroll
  for (int ci = 0; ci < 64; ++ci)
    xv[ci] = x[(((size_t)b * NCH + ci) << 12) + n];
#pragma unroll
  for (int co = 0; co < 8; ++co) {
    float aq = bq[co], ak = bk[co];
#pragma unroll
    for (int ci = 0; ci < 64; ++ci) {
      aq += wq[co * 64 + ci] * xv[ci];
      ak += wk[co * 64 + ci] * xv[ci];
    }
    q[((size_t)(b * 8 + co) << 12) + n] = aq;
    k[((size_t)(b * 8 + co) << 12) + n] = ak;
  }
}

// ---------------------------------------------------------------------------
// PAM pass 1 (flash stats): per 64 rows n, stream m in 64-chunks, S = Q^T K
// via WMMA (K=8 zero-padded to 32 once in LDS), online row max / sum.
// qt: [n][d] f16; ktT: [m][d] f16 (B transposed).
// ---------------------------------------------------------------------------
__global__ void k_pam_stats(const float* __restrict__ q, const float* __restrict__ kmat,
                            float* __restrict__ mstat, float* __restrict__ lstat) {
  __shared__ _Float16 qt[64 * 40];
  __shared__ _Float16 ktT[64 * 40];
  __shared__ float S[64 * 68];
  __shared__ float rmaxA[64], rsumA[64];
  const int tid = threadIdx.x;
  const int b = blockIdx.x >> 6;
  const int n0 = (blockIdx.x & 63) << 6;
  for (int t = tid; t < 64 * 40; t += 256) { qt[t] = (_Float16)0.f; ktT[t] = (_Float16)0.f; }
  if (tid < 64) { rmaxA[tid] = -3.0e38f; rsumA[tid] = 0.f; }
  __syncthreads();
  for (int t = tid; t < 512; t += 256) {   // d<8 region only (rest stays 0)
    const int d = t >> 6, n = t & 63;
    qt[n * 40 + d] = (_Float16)q[((size_t)(b * 8 + d) << 12) + n0 + n];
  }
  __syncthreads();
  const int wave = tid >> 5, lane = tid & 31;
  const int nr = lane & 15, mb2 = (lane >> 4) << 3;
  for (int m0 = 0; m0 < HW; m0 += 64) {
    for (int t = tid; t < 512; t += 256) {
      const int d = t >> 6, m = t & 63;
      ktT[m * 40 + d] = (_Float16)kmat[((size_t)(b * 8 + d) << 12) + m0 + m];
    }
    __syncthreads();
#pragma unroll
    for (int t2 = 0; t2 < 2; ++t2) {
      const int ti = wave * 2 + t2, tr = ti >> 2, tc = ti & 3;
      v16h af = frag_row(&qt[tr * 16 * 40], 40, lane);
      v16h bf = frag_row(&ktT[tc * 16 * 40], 40, lane);
      v8f c = {};
      c = wmma32(af, bf, c);
#pragma unroll
      for (int r = 0; r < 8; ++r)
        S[(tr * 16 + mb2 + r) * 68 + tc * 16 + nr] = c[r];
    }
    __syncthreads();
    if (tid < 64) {
      float lm = -3.0e38f;
      for (int m = 0; m < 64; ++m) lm = fmaxf(lm, S[tid * 68 + m]);
      const float M = fmaxf(rmaxA[tid], lm);
      float s = rsumA[tid] * __expf(rmaxA[tid] - M);
      for (int m = 0; m < 64; ++m) s += __expf(S[tid * 68 + m] - M);
      rmaxA[tid] = M; rsumA[tid] = s;
    }
    __syncthreads();
  }
  if (tid < 64) {
    mstat[((size_t)b << 12) + n0 + tid] = rmaxA[tid];
    lstat[((size_t)b << 12) + n0 + tid] = rsumA[tid];
  }
}

// ---------------------------------------------------------------------------
// PAM pass 2: O[c,n] = sum_m v[c,m] * exp(S[n,m]-max_n).  P written directly
// transposed ([n][m] f16, lane-contiguous in m) and consumed by WMMA.
// ---------------------------------------------------------------------------
__global__ void k_pam_apply(const float* __restrict__ q, const float* __restrict__ kmat,
                            const float* __restrict__ v,
                            const float* __restrict__ mstat, const float* __restrict__ lstat,
                            const float* __restrict__ x, const float* __restrict__ gptr,
                            float* __restrict__ xpam) {
  __shared__ _Float16 qt[64 * 40];
  __shared__ _Float16 ktT[64 * 40];
  __shared__ _Float16 Vt[64 * 72];   // [c][m]
  __shared__ _Float16 PT[64 * 72];   // [n][m]
  __shared__ float rmaxA[64], rinvA[64];
  const int tid = threadIdx.x;
  const int b = blockIdx.x >> 6;
  const int n0 = (blockIdx.x & 63) << 6;
  for (int t = tid; t < 64 * 40; t += 256) { qt[t] = (_Float16)0.f; ktT[t] = (_Float16)0.f; }
  if (tid < 64) {
    rmaxA[tid] = mstat[((size_t)b << 12) + n0 + tid];
    rinvA[tid] = 1.0f / lstat[((size_t)b << 12) + n0 + tid];
  }
  __syncthreads();
  for (int t = tid; t < 512; t += 256) {
    const int d = t >> 6, n = t & 63;
    qt[n * 40 + d] = (_Float16)q[((size_t)(b * 8 + d) << 12) + n0 + n];
  }
  __syncthreads();
  const int wave = tid >> 5, lane = tid & 31;
  const int nr = lane & 15, mb2 = (lane >> 4) << 3;
  const float gamma = gptr[0];
  v8f accO0 = {}, accO1 = {};
  for (int m0 = 0; m0 < HW; m0 += 64) {
    for (int t = tid; t < 512; t += 256) {
      const int d = t >> 6, m = t & 63;
      ktT[m * 40 + d] = (_Float16)kmat[((size_t)(b * 8 + d) << 12) + m0 + m];
    }
    for (int f = tid; f < 1024; f += 256) {   // V tile: 64 x 16 float4
      const int c = f >> 4, m4 = f & 15;
      const float4* p4 =
          (const float4*)(v + (((size_t)b * NCH + c) << 12) + m0) + m4;
      st4h(&Vt[c * 72 + m4 * 4], *p4);
    }
    __syncthreads();
    // phase 1: S tiles -> PT[n][m] = exp(S - rowmax) in f16
#pragma unroll
    for (int t2 = 0; t2 < 2; ++t2) {
      const int ti = wave * 2 + t2, tr = ti >> 2, tc = ti & 3;
      v16h af = frag_row(&qt[tr * 16 * 40], 40, lane);
      v16h bf = frag_row(&ktT[tc * 16 * 40], 40, lane);
      v8f c = {};
      c = wmma32(af, bf, c);
#pragma unroll
      for (int r = 0; r < 8; ++r) {
        const int n = tr * 16 + mb2 + r;
        const int m = tc * 16 + nr;
        PT[n * 72 + m] = (_Float16)__expf(c[r] - rmaxA[n]);
      }
    }
    __syncthreads();
    // phase 2: O += V_tile @ P  (K = 64 over this m-chunk)
#pragma unroll
    for (int t2 = 0; t2 < 2; ++t2) {
      const int ti = wave * 2 + t2, tcO = ti >> 2, tnO = ti & 3;
      v8f a = t2 ? accO1 : accO0;
#pragma unroll
      for (int ks = 0; ks < 64; ks += 32) {
        v16h af = frag_row(&Vt[tcO * 16 * 72 + ks], 72, lane);
        v16h bf = frag_row(&PT[tnO * 16 * 72 + ks], 72, lane);
        a = wmma32(af, bf, a);
      }
      if (t2) accO1 = a; else accO0 = a;
    }
    __syncthreads();
  }
#pragma unroll
  for (int t2 = 0; t2 < 2; ++t2) {
    const int ti = wave * 2 + t2, tcO = ti >> 2, tnO = ti & 3;
    v8f a = t2 ? accO1 : accO0;
#pragma unroll
    for (int r = 0; r < 8; ++r) {
      const int c = tcO * 16 + mb2 + r;
      const int n = tnO * 16 + nr;
      const size_t oi = (((size_t)b * NCH + c) << 12) + n0 + n;
      xpam[oi] = gamma * a[r] * rinvA[n] + x[oi];
    }
  }
}

// ---------------------------------------------------------------------------
// 3x3 conv as implicit GEMM: 9 taps x (64co x 64ci GEMM) per image row.
// W staged [co][ci] f16; shifted input staged transposed [x][ci] f16.
// Optional second input added on load (final conv's atten + x).
// ---------------------------------------------------------------------------
__global__ void k_conv3x3(const float* __restrict__ in, const float* __restrict__ in2,
                          const float* __restrict__ W, const float* __restrict__ bias,
                          const float* __restrict__ bn_g, const float* __restrict__ bn_b,
                          float* __restrict__ out, int act) {
  __shared__ _Float16 ldsW[64 * 72];
  __shared__ _Float16 ldsXT[64 * 72];  // [x][ci]
  const int tid = threadIdx.x;
  const int b = blockIdx.x >> 6;
  const int y = blockIdx.x & 63;
  const int wave = tid >> 5, lane = tid & 31;
  const int nr = lane & 15, mb2 = (lane >> 4) << 3;
  v8f acc0 = {}, acc1 = {};
  for (int ky = 0; ky < 3; ++ky) {
    const int iy = y + ky - 1;
    if (iy < 0 || iy > 63) continue;
    for (int kx = 0; kx < 3; ++kx) {
      for (int t = tid; t < 4096; t += 256) {
        const int co = t >> 6, ci = t & 63;
        ldsW[co * 72 + ci] = (_Float16)W[(size_t)(co * 64 + ci) * 9 + ky * 3 + kx];
      }
      for (int t = tid; t < 4096; t += 256) {
        const int ci = t >> 6, xx = t & 63;
        const int ix = xx + kx - 1;
        float vv = 0.f;
        if (ix >= 0 && ix < 64) {
          const size_t gi = (((size_t)b * NCH + ci) << 12) + iy * 64 + ix;
          vv = in[gi];
          if (in2) vv += in2[gi];
        }
        ldsXT[xx * 72 + ci] = (_Float16)vv;
      }
      __syncthreads();
#pragma unroll
      for (int t2 = 0; t2 < 2; ++t2) {
        const int ti = wave * 2 + t2, tm = ti >> 2, tn = ti & 3;
        v8f a = t2 ? acc1 : acc0;
#pragma unroll
        for (int ks = 0; ks < 64; ks += 32) {
          v16h af = frag_row(&ldsW[tm * 16 * 72 + ks], 72, lane);
          v16h bf = frag_row(&ldsXT[tn * 16 * 72 + ks], 72, lane);
          a = wmma32(af, bf, a);
        }
        if (t2) acc1 = a; else acc0 = a;
      }
      __syncthreads();
    }
  }
#pragma unroll
  for (int t2 = 0; t2 < 2; ++t2) {
    const int ti = wave * 2 + t2, tm = ti >> 2, tn = ti & 3;
    v8f a = t2 ? acc1 : acc0;
#pragma unroll
    for (int r = 0; r < 8; ++r) {
      const int co = tm * 16 + mb2 + r;
      const int xx = tn * 16 + nr;
      float v = a[r] + bias[co];
      if (bn_g) v = v * (bn_g[co] * BN_SCALE) + bn_b[co];
      if (act == 1) v = fmaxf(v, 0.f);
      out[(((size_t)b * NCH + co) << 12) + y * 64 + xx] = v;
    }
  }
}

// ---------------------------------------------------------------------------
// Small VALU kernels.
// ---------------------------------------------------------------------------
__global__ void k_dwconv(const float* __restrict__ in, const float* __restrict__ w,
                         const float* __restrict__ bias,
                         const float* __restrict__ g, const float* __restrict__ bb,
                         float* __restrict__ out) {
  const int gi = blockIdx.x * 256 + threadIdx.x;
  const int b = gi >> 18, c = (gi >> 12) & 63, p = gi & 4095;
  const int y = p >> 6, xx = p & 63;
  float s = bias[c];
#pragma unroll
  for (int ky = 0; ky < 3; ++ky) {
    const int iy = y + ky - 1;
    if (iy < 0 || iy > 63) continue;
#pragma unroll
    for (int kx = 0; kx < 3; ++kx) {
      const int ix = xx + kx - 1;
      if (ix < 0 || ix > 63) continue;
      s += w[c * 9 + ky * 3 + kx] * in[(((size_t)b * NCH + c) << 12) + iy * 64 + ix];
    }
  }
  s = s * (g[c] * BN_SCALE) + bb[c];
  out[gi] = gelu_exact(s);
}

__global__ void k_pool(const float* __restrict__ in, float* __restrict__ out) {
  __shared__ float red[256];
  const int bc = blockIdx.x, tid = threadIdx.x;
  float s = 0.f;
  for (int i = tid; i < 4096; i += 256) s += in[((size_t)bc << 12) + i];
  red[tid] = s; __syncthreads();
  for (int st = 128; st > 0; st >>= 1) {
    if (tid < st) red[tid] += red[tid + st];
    __syncthreads();
  }
  if (tid == 0) out[bc] = red[0] * (1.0f / 4096.0f);
}

__global__ void k_se(const float* __restrict__ pool,
                     const float* __restrict__ w1, const float* __restrict__ b1,
                     const float* __restrict__ g, const float* __restrict__ bb,
                     const float* __restrict__ w2, const float* __restrict__ b2,
                     float* __restrict__ dwci) {
  __shared__ float tt[8];
  const int b = blockIdx.x, tid = threadIdx.x;
  if (tid < 8) {
    float a = b1[tid];
    for (int ci = 0; ci < 64; ++ci) a += w1[tid * 64 + ci] * pool[b * 64 + ci];
    a = a * (g[tid] * BN_SCALE) + bb[tid];
    tt[tid] = gelu_exact(a);
  }
  __syncthreads();
  float a = b2[tid];
#pragma unroll
  for (int j = 0; j < 8; ++j) a += w2[tid * 8 + j] * tt[j];
  dwci[b * 64 + tid] = sigmoidf_(a);
}

__global__ void k_spatial(const float* __restrict__ dw,
                          const float* __restrict__ w1, const float* __restrict__ b1,
                          const float* __restrict__ g, const float* __restrict__ bb,
                          const float* __restrict__ w2, const float* __restrict__ b2,
                          float* __restrict__ dwsi) {
  const int gi = blockIdx.x * 256 + threadIdx.x;
  const int b = gi >> 12, n = gi & 4095;
  float s4[4];
#pragma unroll
  for (int j = 0; j < 4; ++j) {
    float a = b1[j];
    for (int ci = 0; ci < 64; ++ci)
      a += w1[j * 64 + ci] * dw[(((size_t)b * NCH + ci) << 12) + n];
    a = a * (g[j] * BN_SCALE) + bb[j];
    s4[j] = gelu_exact(a);
  }
  float a = b2[0];
#pragma unroll
  for (int j = 0; j < 4; ++j) a += w2[j] * s4[j];
  dwsi[((size_t)b << 12) + n] = sigmoidf_(a);
}

__global__ void k_fuse(const float* __restrict__ cib, const float* __restrict__ sib,
                       const float* __restrict__ dwsi, const float* __restrict__ dwci,
                       float* __restrict__ mixed) {
  const int gi = blockIdx.x * 256 + threadIdx.x;
  const int b = gi >> 18, c = (gi >> 12) & 63, n = gi & 4095;
  mixed[gi] = cib[gi] * dwsi[((size_t)b << 12) + n] + sib[gi] * dwci[b * 64 + c];
}

// ---------------------------------------------------------------------------
extern "C" void kernel_launch(void* const* d_in, const int* in_sizes, int n_in,
                              void* d_out, int out_size, void* d_ws, size_t ws_size,
                              hipStream_t stream) {
  (void)in_sizes; (void)n_in; (void)out_size; (void)ws_size;
  const float* x         = (const float*)d_in[0];
  const float* gamma_cam = (const float*)d_in[1];
  const float* gamma_pam = (const float*)d_in[2];
  const float* pam_wq = (const float*)d_in[3];
  const float* pam_bq = (const float*)d_in[4];
  const float* pam_wk = (const float*)d_in[5];
  const float* pam_bk = (const float*)d_in[6];
  const float* pam_wv = (const float*)d_in[7];
  const float* pam_bv = (const float*)d_in[8];
  const float* b1_w1 = (const float*)d_in[9];
  const float* b1_b1 = (const float*)d_in[10];
  const float* b1_w2 = (const float*)d_in[11];
  const float* b1_b2 = (const float*)d_in[12];
  const float* b2_w1 = (const float*)d_in[13];
  const float* b2_b1 = (const float*)d_in[14];
  const float* b2_w2 = (const float*)d_in[15];
  const float* b2_b2 = (const float*)d_in[16];
  const float* dw_w1 = (const float*)d_in[17];
  const float* dw_b1 = (const float*)d_in[18];
  const float* dw_w2 = (const float*)d_in[19];
  const float* dw_b2 = (const float*)d_in[20];
  const float* dw_g  = (const float*)d_in[21];
  const float* dw_bb = (const float*)d_in[22];
  const float* ci_w1 = (const float*)d_in[23];
  const float* ci_b1 = (const float*)d_in[24];
  const float* ci_w2 = (const float*)d_in[25];
  const float* ci_b2 = (const float*)d_in[26];
  const float* ci_g  = (const float*)d_in[27];
  const float* ci_bb = (const float*)d_in[28];
  const float* si_w1 = (const float*)d_in[29];
  const float* si_b1 = (const float*)d_in[30];
  const float* si_w2 = (const float*)d_in[31];
  const float* si_b2 = (const float*)d_in[32];
  const float* si_g  = (const float*)d_in[33];
  const float* si_bb = (const float*)d_in[34];
  const float* at_w  = (const float*)d_in[35];
  const float* at_b  = (const float*)d_in[36];
  const float* at_g  = (const float*)d_in[37];
  const float* at_bb = (const float*)d_in[38];
  const float* f_w   = (const float*)d_in[39];
  const float* f_b   = (const float*)d_in[40];
  const float* f_g   = (const float*)d_in[41];
  const float* f_bb  = (const float*)d_in[42];

  const size_t BCHW = (size_t)4 * 64 * 4096;
  float* w = (float*)d_ws;
  size_t off = 0;
  auto alloc = [&](size_t n) { float* p = w + off; off += n; return p; };
  float* attc  = alloc(4 * 64 * 64);
  float* qb    = alloc(4 * 8 * 4096);
  float* kb    = alloc(4 * 8 * 4096);
  float* vb    = alloc(BCHW);
  float* mst   = alloc(4 * 4096);
  float* lst   = alloc(4 * 4096);
  float* xcam  = alloc(BCHW);
  float* xpam  = alloc(BCHW);
  float* t1    = alloc(BCHW);      // reused: basic-block temp, then `mixed`
  float* cib   = alloc(BCHW);
  float* sib   = alloc(BCHW);
  float* dwt   = alloc(BCHW);      // reused: dw1 output, then `atten`
  float* dwb   = alloc(BCHW);
  float* poolb = alloc(256);
  float* dwci  = alloc(256);
  float* dwsi  = alloc(4 * 4096);

  const dim3 blk(256);
  // CAM
  k_cam_energy<<<4, blk, 0, stream>>>(x, attc);
  k_gemm64<<<256, blk, 0, stream>>>(attc, 4096, 64, x, nullptr, nullptr, nullptr,
                                    nullptr, x, gamma_cam, xcam, 0);
  // PAM
  k_gemm64<<<256, blk, 0, stream>>>(pam_wv, 0, 64, x, nullptr, pam_bv, nullptr,
                                    nullptr, nullptr, nullptr, vb, 0);
  k_pam_qk<<<64, blk, 0, stream>>>(x, pam_wq, pam_bq, pam_wk, pam_bk, qb, kb);
  k_pam_stats<<<256, blk, 0, stream>>>(qb, kb, mst, lst);
  k_pam_apply<<<256, blk, 0, stream>>>(qb, kb, vb, mst, lst, x, gamma_pam, xpam);
  // basic dual blocks
  k_gemm64<<<256, blk, 0, stream>>>(b1_w1, 0, 64, xcam, nullptr, b1_b1, nullptr,
                                    nullptr, nullptr, nullptr, t1, 1);
  k_conv3x3<<<256, blk, 0, stream>>>(t1, nullptr, b1_w2, b1_b2, nullptr, nullptr, cib, 1);
  k_gemm64<<<256, blk, 0, stream>>>(b2_w1, 0, 64, xpam, nullptr, b2_b1, nullptr,
                                    nullptr, nullptr, nullptr, t1, 1);
  k_conv3x3<<<256, blk, 0, stream>>>(t1, nullptr, b2_w2, b2_b2, nullptr, nullptr, sib, 1);
  // dw path on virtual concat [x_cam; x_pam]
  k_gemm64<<<256, blk, 0, stream>>>(dw_w1, 0, 128, xcam, xpam, dw_b1, nullptr,
                                    nullptr, nullptr, nullptr, dwt, 0);
  k_dwconv<<<4096, blk, 0, stream>>>(dwt, dw_w2, dw_b2, dw_g, dw_bb, dwb);
  // SE + spatial gates
  k_pool<<<256, blk, 0, stream>>>(dwb, poolb);
  k_se<<<4, dim3(64), 0, stream>>>(poolb, ci_w1, ci_b1, ci_g, ci_bb, ci_w2, ci_b2, dwci);
  k_spatial<<<64, blk, 0, stream>>>(dwb, si_w1, si_b1, si_g, si_bb, si_w2, si_b2, dwsi);
  k_fuse<<<4096, blk, 0, stream>>>(cib, sib, dwsi, dwci, t1);
  // attention conv + final conv (input = atten + x, fused on load)
  k_conv3x3<<<256, blk, 0, stream>>>(t1, nullptr, at_w, at_b, at_g, at_bb, dwt, 1);
  k_conv3x3<<<256, blk, 0, stream>>>(dwt, x, f_w, f_b, f_g, f_bb, (float*)d_out, 1);
}
